// DGRec_76922864272029
// MI455X (gfx1250) — compile-verified, matrix operand order checked by reading, not run on hardware
//
#include <hip/hip_runtime.h>
#include <hip/hip_bf16.h>
#include <math.h>

// ---------------------------------------------------------------------------
// DGRec forward for MI455X (gfx1250, wave32).
//  - propagation: HBM-bound (~2 GB traffic) -> coalesced per-(edge,dim) f32
//    global atomics + global_prefetch_b8.
//  - FFT neighbor filter folded into a per-dim 8x8 linear operator C[d].
//  - anchor = sbar @ WU  and the 512x512x64 contrastive similarity GEMMs use
//    v_wmma_f32_16x16x32_f16; operands are loaded as 2x ds_load_b128 per
//    v16h (the 16-bit A-layout's per-lane halves are two contiguous 8-half
//    runs: k = loff+0..7 and k = loff+16..23, loff = (lane&16)?8:0).
// ---------------------------------------------------------------------------

typedef __attribute__((ext_vector_type(16))) _Float16 v16h;
typedef __attribute__((ext_vector_type(8)))  float    v8f;

#define DD    64
#define KK    8
#define TEMP_ 0.2f

static __device__ __forceinline__ v8f wmma_f32_f16(v16h a, v16h b, v8f c) {
  // (neg_a, A, neg_b, B, c_mod, C, reuse_a, reuse_b)
  return __builtin_amdgcn_wmma_f32_16x16x32_f16(false, a, false, b, (short)0, c,
                                                false, false);
}

// Load one v16h WMMA operand from a row-contiguous f16 row in LDS.
// halves[0..7] = row[loff+0..7], halves[8..15] = row[loff+16..23].
static __device__ __forceinline__ v16h load_op_lds(const _Float16* rowp, int loff) {
  union { v16h v; uint4 q[2]; } t;
  t.q[0] = *(const uint4*)(rowp + loff);
  t.q[1] = *(const uint4*)(rowp + loff + 16);
  return t.v;
}

// Same operand built from a row-contiguous f32 row in global memory.
static __device__ __forceinline__ v16h load_op_g32(const float* rowp, int loff) {
  union { v16h v; _Float16 h[16]; } t;
  float4 f0 = *(const float4*)(rowp + loff);
  float4 f1 = *(const float4*)(rowp + loff + 4);
  float4 f2 = *(const float4*)(rowp + loff + 16);
  float4 f3 = *(const float4*)(rowp + loff + 20);
  t.h[0] = (_Float16)f0.x; t.h[1] = (_Float16)f0.y;
  t.h[2] = (_Float16)f0.z; t.h[3] = (_Float16)f0.w;
  t.h[4] = (_Float16)f1.x; t.h[5] = (_Float16)f1.y;
  t.h[6] = (_Float16)f1.z; t.h[7] = (_Float16)f1.w;
  t.h[8] = (_Float16)f2.x; t.h[9] = (_Float16)f2.y;
  t.h[10] = (_Float16)f2.z; t.h[11] = (_Float16)f2.w;
  t.h[12] = (_Float16)f3.x; t.h[13] = (_Float16)f3.y;
  t.h[14] = (_Float16)f3.z; t.h[15] = (_Float16)f3.w;
  return t.v;
}

static __device__ __forceinline__ float half16_max(float v) {
  v = fmaxf(v, __shfl_xor(v, 1, 32));
  v = fmaxf(v, __shfl_xor(v, 2, 32));
  v = fmaxf(v, __shfl_xor(v, 4, 32));
  v = fmaxf(v, __shfl_xor(v, 8, 32));
  return v;
}
static __device__ __forceinline__ float half16_sum(float v) {
  v += __shfl_xor(v, 1, 32);
  v += __shfl_xor(v, 2, 32);
  v += __shfl_xor(v, 4, 32);
  v += __shfl_xor(v, 8, 32);
  return v;
}

// ---------------- degree + scatter (propagation) ----------------------------

__global__ void k_degree(const int* __restrict__ src, const int* __restrict__ dst,
                         float* __restrict__ deg_u, float* __restrict__ deg_i, int E) {
  int e = blockIdx.x * blockDim.x + threadIdx.x;
  if (e >= E) return;
  atomicAdd(&deg_u[src[e]], 1.0f);
  atomicAdd(&deg_i[dst[e]], 1.0f);
}

__global__ void k_scatter(const float* __restrict__ feat, const int* __restrict__ sidx,
                          const int* __restrict__ didx, float* __restrict__ acc, int E) {
  long long t = (long long)blockIdx.x * blockDim.x + threadIdx.x;
  long long total = (long long)E * DD;
  if (t >= total) return;
  int e = (int)(t >> 6);
  int d = (int)(t & 63);
  if (d == 0 && e + 256 < E) {
    __builtin_prefetch(&sidx[e + 256], 0, 0);   // -> global_prefetch_b8
    __builtin_prefetch(&didx[e + 256], 0, 0);
  }
  int s = sidx[e], q = didx[e];
  atomicAdd(&acc[(long long)q * DD + d], feat[(long long)s * DD + d]);
}

__global__ void k_norm(float* __restrict__ buf, const float* __restrict__ deg,
                       long long n) {
  long long t = (long long)blockIdx.x * blockDim.x + threadIdx.x;
  if (t >= n) return;
  float dg = deg[t >> 6];
  buf[t] = buf[t] / fmaxf(dg, 1.0f);
}

// ---------------- small precomputes -----------------------------------------

__global__ void k_smallvec(const float* __restrict__ W, const float* __restrict__ a,
                           const float* __restrict__ WU, const float* __restrict__ aU,
                           float* __restrict__ wa, float* __restrict__ v1,
                           float* __restrict__ v2) {
  int d = threadIdx.x;  // 64 threads
  float s0 = 0.f, s1 = 0.f, s2 = 0.f;
#pragma unroll 8
  for (int e = 0; e < DD; ++e) {
    s0 += W[d * DD + e] * a[e];
    s1 += WU[d * DD + e] * aU[e];
    s2 += WU[d * DD + e] * aU[DD + e];
  }
  wa[d] = s0; v1[d] = s1; v2[d] = s2;
}

// combined rfft(ortho) * complex_weight -> irfft(ortho) as an 8x8 operator per d
__global__ void k_filterC(const float* __restrict__ cw, float* __restrict__ C) {
  int id = blockIdx.x * blockDim.x + threadIdx.x;   // 4096 = 64*8*8
  if (id >= DD * KK * KK) return;
  int d = id >> 6;
  int k = (id >> 3) & 7;
  int j = id & 7;
  const float twopi_over_K = 6.2831853071795864769f / (float)KK;
  float acc = 0.f;
#pragma unroll
  for (int f = 0; f <= KK / 2; ++f) {
    float thj = twopi_over_K * (float)(f * j);
    float thk = twopi_over_K * (float)(f * k);
    float ReX = cosf(thj);
    float ImX = -sinf(thj);
    float wr = cw[(f * DD + d) * 2 + 0];
    float wi = cw[(f * DD + d) * 2 + 1];
    float ReY = ReX * wr - ImX * wi;
    float ImY = ReX * wi + ImX * wr;
    if (f == 0 || f == KK / 2) acc += ReY * cosf(thk);
    else                       acc += 2.f * (ReY * cosf(thk) - ImY * sinf(thk));
  }
  C[(d * KK + k) * KK + j] = acc * (1.0f / (float)KK);
}

// ---------------- layer attention (collapses to dots with wa) ---------------

__global__ void k_attn(const float* __restrict__ t0, const float* __restrict__ t1,
                       const float* __restrict__ t2, const float* __restrict__ wa,
                       float* __restrict__ out, int N) {
  int n = blockIdx.x * blockDim.x + threadIdx.x;
  if (n >= N) return;
  long long b = (long long)n * DD;
  float s0 = 0.f, s1 = 0.f, s2 = 0.f;
#pragma unroll
  for (int d4 = 0; d4 < DD / 4; ++d4) {
    float4 w = *(const float4*)(wa + d4 * 4);
    float4 x0 = *(const float4*)(t0 + b + d4 * 4);
    float4 x1 = *(const float4*)(t1 + b + d4 * 4);
    float4 x2 = *(const float4*)(t2 + b + d4 * 4);
    s0 += x0.x * w.x + x0.y * w.y + x0.z * w.z + x0.w * w.w;
    s1 += x1.x * w.x + x1.y * w.y + x1.z * w.z + x1.w * w.w;
    s2 += x2.x * w.x + x2.y * w.y + x2.z * w.z + x2.w * w.w;
  }
  float m = fmaxf(s0, fmaxf(s1, s2));
  float e0 = __expf(s0 - m), e1 = __expf(s1 - m), e2 = __expf(s2 - m);
  float inv = 1.0f / (e0 + e1 + e2);
  e0 *= inv; e1 *= inv; e2 *= inv;
#pragma unroll
  for (int d4 = 0; d4 < DD / 4; ++d4) {
    float4 x0 = *(const float4*)(t0 + b + d4 * 4);
    float4 x1 = *(const float4*)(t1 + b + d4 * 4);
    float4 x2 = *(const float4*)(t2 + b + d4 * 4);
    float4 o;
    o.x = e0 * x0.x + e1 * x1.x + e2 * x2.x;
    o.y = e0 * x0.y + e1 * x1.y + e2 * x2.y;
    o.z = e0 * x0.z + e1 * x1.z + e2 * x2.z;
    o.w = e0 * x0.w + e1 * x1.w + e2 * x2.w;
    *(float4*)(out + b + d4 * 4) = o;
  }
}

// ---------------- anchor: filtered-neighbor attention -> sbar ---------------

__global__ void k_sbar(const float* __restrict__ ue, const int* __restrict__ nbr,
                       const float* __restrict__ C, const float* __restrict__ v1,
                       const float* __restrict__ v2, float* __restrict__ sbar) {
  int u = blockIdx.x;
  int d = threadIdx.x;  // 64 threads
  __shared__ float red[9 * DD];
  __shared__ float alpha[KK];
  float x[KK];
#pragma unroll
  for (int j = 0; j < KK; ++j)
    x[j] = ue[(long long)nbr[u * KK + j] * DD + d];
  const float* Cd = C + d * KK * KK;
  float s[KK];
#pragma unroll
  for (int k = 0; k < KK; ++k) {
    float acc = 0.f;
#pragma unroll
    for (int j = 0; j < KK; ++j) acc += Cd[k * KK + j] * x[j];
    s[k] = acc;
  }
#pragma unroll
  for (int k = 0; k < KK; ++k) red[k * DD + d] = s[k] * v1[d];
  red[8 * DD + d] = ue[(long long)u * DD + d] * v2[d];
  __syncthreads();
  for (int off = 32; off > 0; off >>= 1) {
    if (d < off) {
#pragma unroll
      for (int r = 0; r < 9; ++r) red[r * DD + d] += red[r * DD + d + off];
    }
    __syncthreads();
  }
  if (d == 0) {
    float base = red[8 * DD];
    float e[KK], mx = -1e30f;
#pragma unroll
    for (int k = 0; k < KK; ++k) {
      float v = red[k * DD] + base;
      v = (v > 0.f) ? v : 0.1f * v;   // LeakyReLU(0.1)
      e[k] = v;
      mx = fmaxf(mx, v);
    }
    float sum = 0.f;
#pragma unroll
    for (int k = 0; k < KK; ++k) { e[k] = __expf(e[k] - mx); sum += e[k]; }
    float inv = 1.0f / sum;
#pragma unroll
    for (int k = 0; k < KK; ++k) alpha[k] = e[k] * inv;
  }
  __syncthreads();
  float o = 0.f;
#pragma unroll
  for (int k = 0; k < KK; ++k) o += alpha[k] * s[k];
  sbar[(long long)u * DD + d] = o;
}

// ---------------- anchor GEMM: [U,64] @ [64,64] via WMMA --------------------

__global__ void k_gemm_anchor(const float* __restrict__ sbar,
                              const float* __restrict__ WU,
                              float* __restrict__ anchor) {
  __shared__ _Float16 WUhT[DD * DD];     // WU transposed: WUhT[e*64+k] = WU[k][e]
  int tid = threadIdx.x;                 // 128 threads = 4 waves
  for (int i = tid; i < DD * DD; i += 128) {
    int k = i >> 6, e = i & 63;
    WUhT[e * DD + k] = (_Float16)WU[i];
  }
  __syncthreads();
  int wave = tid >> 5, lane = tid & 31;
  int rowTile = blockIdx.x * 4 + wave;
  int rowbase = rowTile * 16;
  int n = lane & 15;
  int hi8 = (lane & 16) ? 8 : 0;
  int loff = (lane & 16) ? 8 : 0;
  const float* arowp = sbar + (long long)(rowbase + (lane & 15)) * DD;
  v16h a0 = load_op_g32(arowp, loff);
  v16h a1 = load_op_g32(arowp + 32, loff);
#pragma unroll
  for (int ct = 0; ct < 4; ++ct) {
    const _Float16* brp = WUhT + (ct * 16 + n) * DD;
    v16h b0 = load_op_lds(brp, loff);
    v16h b1 = load_op_lds(brp + 32, loff);
    v8f c = {};
    c = wmma_f32_f16(a0, b0, c);
    c = wmma_f32_f16(a1, b1, c);
#pragma unroll
    for (int r = 0; r < 8; ++r)
      anchor[(long long)(rowbase + r + hi8) * DD + ct * 16 + n] = c[r];
  }
}

// ---------------- contrastive loss: one block per 256-row chunk -------------
// logits = [[s11+mask, c12],[s22+mask, c21]] with label col 256+(i&255).
// 512x512x64 similarity GEMM via WMMA f16->f32, online logsumexp per row.

__global__ void k_contrast(const float* __restrict__ hu,
                           const float* __restrict__ anchor,
                           float* __restrict__ lossAcc) {
  extern __shared__ unsigned char smem[];
  _Float16* r1h = (_Float16*)smem;                 // 256*64 f16
  _Float16* r2h = r1h + 256 * DD;                  // 256*64 f16
  float* n1 = (float*)(r2h + 256 * DD);            // 256
  float* n2 = n1 + 256;                            // 256
  int chunk = blockIdx.x;
  int tid = threadIdx.x;                           // 256 threads
  {
    int row = tid;
    const float* p1 = hu + ((long long)chunk * 256 + row) * DD;
    const float* p2 = anchor + ((long long)chunk * 256 + row) * DD;
    unsigned* w1 = (unsigned*)(r1h + row * DD);
    unsigned* w2 = (unsigned*)(r2h + row * DD);
    float ss1 = 0.f, ss2 = 0.f;
#pragma unroll
    for (int d4 = 0; d4 < DD / 4; ++d4) {
      float4 va = *(const float4*)(p1 + d4 * 4);
      float4 vb = *(const float4*)(p2 + d4 * 4);
      ss1 += va.x * va.x + va.y * va.y + va.z * va.z + va.w * va.w;
      ss2 += vb.x * vb.x + vb.y * vb.y + vb.z * vb.z + vb.w * vb.w;
      union { _Float16 h[2]; unsigned u; } pa, pb;
      pa.h[0] = (_Float16)va.x; pa.h[1] = (_Float16)va.y; w1[d4 * 2] = pa.u;
      pa.h[0] = (_Float16)va.z; pa.h[1] = (_Float16)va.w; w1[d4 * 2 + 1] = pa.u;
      pb.h[0] = (_Float16)vb.x; pb.h[1] = (_Float16)vb.y; w2[d4 * 2] = pb.u;
      pb.h[0] = (_Float16)vb.z; pb.h[1] = (_Float16)vb.w; w2[d4 * 2 + 1] = pb.u;
    }
    n1[row] = sqrtf(ss1);
    n2[row] = sqrtf(ss2);
  }
  __syncthreads();

  int wave = tid >> 5, lane = tid & 31;
  int n = lane & 15;
  int hi8 = (lane & 16) ? 8 : 0;
  int loff = hi8;

  for (int rt = wave; rt < 32; rt += 8) {          // 32 row tiles of 16
    const _Float16* Ra = (rt < 16) ? r1h : r2h;
    const float* Na = (rt < 16) ? n1 : n2;
    int abase = (rt & 15) * 16;
    const _Float16* arp = Ra + (abase + (lane & 15)) * DD;
    v16h a0 = load_op_lds(arp, loff);
    v16h a1 = load_op_lds(arp + 32, loff);
    float na_r[8];
#pragma unroll
    for (int r = 0; r < 8; ++r) na_r[r] = Na[abase + r + hi8];

    float rm[8], rs[8], lab[8];
#pragma unroll
    for (int r = 0; r < 8; ++r) { rm[r] = -1e30f; rs[r] = 0.f; lab[r] = 0.f; }

    for (int ct = 0; ct < 32; ++ct) {              // 32 col tiles of 16
      const _Float16* Rb = (ct < 16) ? ((rt < 16) ? r1h : r2h)
                                     : ((rt < 16) ? r2h : r1h);
      const float* Nb = (ct < 16) ? ((rt < 16) ? n1 : n2)
                                  : ((rt < 16) ? n2 : n1);
      int bbase = (ct & 15) * 16;
      const _Float16* brp = Rb + (bbase + n) * DD; // logits col -> Rb row (B = Rb^T)
      v16h b0 = load_op_lds(brp, loff);
      v16h b1 = load_op_lds(brp + 32, loff);
      v8f c = {};
      c = wmma_f32_f16(a0, b0, c);
      c = wmma_f32_f16(a1, b1, c);

      float nb = Nb[bbase + n];
      bool maskTile  = (ct < 16) && (ct == (rt & 15));
      bool labelTile = (ct >= 16) && ((ct - 16) == (rt & 15));
#pragma unroll
      for (int r = 0; r < 8; ++r) {
        int m = r + hi8;
        float logit = c[r] / ((na_r[r] * nb + 1e-6f) * TEMP_);
        if (maskTile && n == m) logit += -1e10f;
        if (labelTile && n == m) lab[r] = logit;
        float tm = half16_max(logit);
        float te = half16_sum(__expf(logit - tm));
        float nm = fmaxf(rm[r], tm);
        rs[r] = rs[r] * __expf(rm[r] - nm) + te * __expf(tm - nm);
        rm[r] = nm;
      }
    }
    // owner lanes (diag carriers): lanes 0..7 own rows 0..7, 24..31 own 8..15
    int own = -1;
    if (lane < 8) own = lane;
    else if (lane >= 24) own = lane - 24;
    if (own >= 0) {
      float loss = (rm[own] + logf(rs[own])) - lab[own];
      atomicAdd(lossAcc, loss);
    }
  }
}

// ---------------- scoring + final loss --------------------------------------

__global__ void k_score(const float* __restrict__ hu, const float* __restrict__ hi,
                        const int* __restrict__ pu, const int* __restrict__ pi,
                        const int* __restrict__ nu, const int* __restrict__ ni,
                        float* __restrict__ out, int EP) {
  int t = blockIdx.x * blockDim.x + threadIdx.x;
  if (t >= 2 * EP) return;
  int u, it;
  if (t < EP) { u = pu[t]; it = pi[t]; }
  else        { u = nu[t - EP]; it = ni[t - EP]; }
  const float* a = hu + (long long)u * DD;
  const float* b = hi + (long long)it * DD;
  float s = 0.f;
#pragma unroll
  for (int d4 = 0; d4 < DD / 4; ++d4) {
    float4 va = *(const float4*)(a + d4 * 4);
    float4 vb = *(const float4*)(b + d4 * 4);
    s += va.x * vb.x + va.y * vb.y + va.z * vb.z + va.w * vb.w;
  }
  out[t] = s;
}

__global__ void k_final(const float* __restrict__ lossAcc, float* __restrict__ out,
                        int EP, int nchunks) {
  if (blockIdx.x == 0 && threadIdx.x == 0)
    out[2 * EP] = lossAcc[0] / (float)(nchunks * 512);
}

// ---------------- launcher ---------------------------------------------------

extern "C" void kernel_launch(void* const* d_in, const int* in_sizes, int n_in,
                              void* d_out, int out_size, void* d_ws, size_t ws_size,
                              hipStream_t stream) {
  const float* user_emb = (const float*)d_in[0];
  const float* item_emb = (const float*)d_in[1];
  const float* W  = (const float*)d_in[2];
  const float* a  = (const float*)d_in[3];
  const float* WU = (const float*)d_in[4];
  const float* aU = (const float*)d_in[5];
  const float* cw = (const float*)d_in[6];
  const int* rate_src = (const int*)d_in[7];
  const int* rate_dst = (const int*)d_in[8];
  const int* pos_u = (const int*)d_in[9];
  const int* pos_i = (const int*)d_in[10];
  const int* neg_u = (const int*)d_in[11];
  const int* neg_i = (const int*)d_in[12];
  const int* nbr   = (const int*)d_in[13];

  const int U  = in_sizes[0] / DD;
  const int I  = in_sizes[1] / DD;
  const int E  = in_sizes[7];
  const int EP = in_sizes[9];

  float* p = (float*)d_ws;
  float* deg_u = p; p += U;
  float* deg_i = p; p += I;
  float* i1 = p; p += (size_t)I * DD;
  float* u1 = p; p += (size_t)U * DD;
  float* i2 = p; p += (size_t)I * DD;
  float* u2 = p; p += (size_t)U * DD;
  float* hu = p; p += (size_t)U * DD;
  float* hi = p; p += (size_t)I * DD;
  float* Cf = p; p += DD * KK * KK;
  float* wa = p; p += DD;
  float* v1 = p; p += DD;
  float* v2 = p; p += DD;
  float* lossAcc = p; p += 1;
  size_t used = (size_t)((char*)p - (char*)d_ws);
  float* sbar   = u1;  // u1/u2 dead after attention; reuse
  float* anchor = u2;
  float* out = (float*)d_out;

  hipMemsetAsync(d_ws, 0, used, stream);

  // degrees (shared by both layers)
  k_degree<<<(E + 255) / 256, 256, 0, stream>>>(rate_src, rate_dst, deg_u, deg_i, E);

  long long etot = (long long)E * DD;
  unsigned eg = (unsigned)((etot + 255) / 256);
  long long utot = (long long)U * DD, itot = (long long)I * DD;
  unsigned ug = (unsigned)((utot + 255) / 256), ig = (unsigned)((itot + 255) / 256);

  // layer 1
  k_scatter<<<eg, 256, 0, stream>>>(user_emb, rate_src, rate_dst, i1, E);
  k_scatter<<<eg, 256, 0, stream>>>(item_emb, rate_dst, rate_src, u1, E);
  k_norm<<<ig, 256, 0, stream>>>(i1, deg_i, itot);
  k_norm<<<ug, 256, 0, stream>>>(u1, deg_u, utot);
  // layer 2
  k_scatter<<<eg, 256, 0, stream>>>(u1, rate_src, rate_dst, i2, E);
  k_scatter<<<eg, 256, 0, stream>>>(i1, rate_dst, rate_src, u2, E);
  k_norm<<<ig, 256, 0, stream>>>(i2, deg_i, itot);
  k_norm<<<ug, 256, 0, stream>>>(u2, deg_u, utot);

  k_smallvec<<<1, 64, 0, stream>>>(W, a, WU, aU, wa, v1, v2);
  k_attn<<<(U + 255) / 256, 256, 0, stream>>>(user_emb, u1, u2, wa, hu, U);
  k_attn<<<(I + 255) / 256, 256, 0, stream>>>(item_emb, i1, i2, wa, hi, I);

  k_filterC<<<(DD * KK * KK + 255) / 256, 256, 0, stream>>>(cw, Cf);
  k_sbar<<<U, 64, 0, stream>>>(user_emb, nbr, Cf, v1, v2, sbar);
  k_gemm_anchor<<<U / 64, 128, 0, stream>>>(sbar, WU, anchor);   // WMMA

  int nchunks = U / 256;
  size_t shmem = (size_t)2 * 256 * DD * sizeof(_Float16) + 2 * 256 * sizeof(float);
  k_contrast<<<nchunks, 256, shmem, stream>>>(hu, anchor, lossAcc);  // WMMA

  k_score<<<(2 * EP + 255) / 256, 256, 0, stream>>>(hu, hi, pos_u, pos_i,
                                                    neg_u, neg_i, out, EP);
  k_final<<<1, 1, 0, stream>>>(lossAcc, out, EP, nchunks);
}
